// DsDTW_8349416423710
// MI455X (gfx1250) — matrix operand order repaired and checked by reading, not compiled, not run on hardware
//
#include <hip/hip_runtime.h>

// ---------------------------------------------------------------------------
// DsDTW forward for MI455X (gfx1250): conv+pool -> softDTW bias -> 1-layer
// transformer encoder -> linear(16).  Dense GEMMs use v_wmma_f32_16x16x32_f16
// (f16 in, f32 accumulate); heavy GEMMs compute a 16x64 strip per wave32 so
// each A-fragment feeds 4 WMMAs.  Softmax stages rows via async LDS loads
// (global_load_async_to_lds_b128 + s_wait_asynccnt).  SoftDTW runs one
// workgroup per batch sample with the anti-diagonal wavefront held in LDS.
// ---------------------------------------------------------------------------

typedef __attribute__((ext_vector_type(16))) _Float16 v16h;
typedef __attribute__((ext_vector_type(8)))  _Float16 v8h;
typedef __attribute__((ext_vector_type(8)))  float    v8f;
typedef int v4i_t __attribute__((vector_size(16)));   // GCC-style int4 for async builtin

constexpr int BB  = 32;
constexpr int CIN = 12;
constexpr int TT  = 4095;
constexpr int DM  = 128;
constexpr int LL  = 1024;
constexpr float BIGF = 1e30f;

#define HAS_ASYNC_LDS __has_builtin(__builtin_amdgcn_global_load_async_to_lds_b128)

// ---------------- WMMA fragment helpers (ISA 7.12.2 layouts) ----------------
__device__ inline v16h load_frag_a(const _Float16* __restrict__ A, long long lda,
                                   int row0, int k0, int lane) {
  int m = lane & 15, hi = lane >> 4;
  const _Float16* p = A + (size_t)(row0 + m) * lda + k0 + 8 * hi;
  v8h lo = *(const v8h*)p;
  v8h hh = *(const v8h*)(p + 16);
  v16h r;
#pragma unroll
  for (int i = 0; i < 8; i++) { r[i] = lo[i]; r[i + 8] = hh[i]; }
  return r;
}

__device__ inline v16h load_frag_b(const _Float16* __restrict__ Y, long long ldy,
                                   int col0, int k0, int lane) {
  int n = lane & 15, hi = lane >> 4;
  const _Float16* p = Y + (size_t)(col0 + n) * ldy + k0 + 8 * hi;
  v8h lo = *(const v8h*)p;
  v8h hh = *(const v8h*)(p + 16);
  v16h r;
#pragma unroll
  for (int i = 0; i < 8; i++) { r[i] = lo[i]; r[i + 8] = hh[i]; }
  return r;
}

__device__ inline v8f wmma32(v16h a, v16h b, v8f c) {
  return __builtin_amdgcn_wmma_f32_16x16x32_f16(false, a, false, b, (short)0, c,
                                                false, false);
}

// ---------------- 1) conv1d(k=4,p=2) + avgpool4 + relu + norms --------------
__global__ void conv_kernel(const float* __restrict__ x,
                            const float* __restrict__ cw,
                            const float* __restrict__ cb,
                            const float* __restrict__ mask,
                            float* __restrict__ h, _Float16* __restrict__ h16,
                            float* __restrict__ norms) {
  __shared__ float xs[CIN * 7];
  __shared__ float red[DM];
  int bl = blockIdx.x;          // b*LL + l
  int b  = bl >> 10, l = bl & (LL - 1);
  int d  = threadIdx.x;
  for (int idx = threadIdx.x; idx < CIN * 7; idx += DM) {
    int c = idx / 7, o = idx % 7;
    int t = 4 * l - 2 + o;
    xs[idx] = (t >= 0 && t < TT) ? x[((size_t)b * CIN + c) * TT + t] : 0.f;
  }
  __syncthreads();
  const float* wd = cw + d * CIN * 4;
  float acc = 0.f;
#pragma unroll
  for (int q = 0; q < 4; q++)
#pragma unroll
    for (int c = 0; c < CIN; c++)
#pragma unroll
      for (int k = 0; k < 4; k++) acc += xs[c * 7 + q + k] * wd[c * 4 + k];
  float hv = fmaxf(acc * 0.25f + cb[d], 0.f) * mask[bl];
  h[(size_t)bl * DM + d]   = hv;
  h16[(size_t)bl * DM + d] = (_Float16)hv;
  red[d] = hv * hv;
  __syncthreads();
  for (int s = 64; s > 0; s >>= 1) {
    if (d < s) red[d] += red[d + s];
    __syncthreads();
  }
  if (d == 0) norms[bl] = red[0];
}

// -------- 2) pairwise sq-distance vs sample 0 (WMMA, 16x64 per wave) --------
__global__ void dist_kernel(const _Float16* __restrict__ h16,
                            const float* __restrict__ norms,
                            float* __restrict__ R) {
  int lane = threadIdx.x & 31, wave = threadIdx.x >> 5;
  int b = blockIdx.z;
  int tile = blockIdx.x * 8 + wave;           // 64 i-tiles x 16 j-groups
  int i0 = (tile >> 4) << 4, j0 = (tile & 15) << 6;
  const _Float16* A = h16;                    // anchor = batch 0
  const _Float16* Y = h16 + (size_t)b * LL * DM;
  v8f c[4];
#pragma unroll
  for (int g = 0; g < 4; g++)
    for (int i = 0; i < 8; i++) c[g][i] = 0.f;
#pragma unroll
  for (int k = 0; k < DM; k += 32) {
    v16h a = load_frag_a(A, DM, i0, k, lane);
#pragma unroll
    for (int g = 0; g < 4; g++)
      c[g] = wmma32(a, load_frag_b(Y, DM, j0 + 16 * g, k, lane), c[g]);
  }
  int nn = lane & 15, hi = lane >> 4;
#pragma unroll
  for (int g = 0; g < 4; g++) {
    int col = j0 + 16 * g + nn;
    float nb = norms[(size_t)b * LL + col];
#pragma unroll
    for (int r = 0; r < 8; r++) {
      int row = i0 + r + 8 * hi;
      R[((size_t)b * LL + row) * LL + col] = norms[row] + nb - 2.f * c[g][r];
    }
  }
}

// ---------------- 3) soft-DTW wavefront, R in-place, per-batch min/max ------
__global__ void __launch_bounds__(1024, 1)
softdtw_kernel(float* __restrict__ R, float* __restrict__ rmin,
               float* __restrict__ rmax) {
  __shared__ float sb[3][LL];
  __shared__ float red[LL];
  int b = blockIdx.x, i = threadIdx.x;
  sb[0][i] = BIGF; sb[1][i] = BIGF;
  __syncthreads();
  int c2 = 0, c1 = 1, cc = 2;
  float lmin = 3e38f, lmax = -3e38f;
  for (int k = 0; k < 2 * LL - 1; k++) {
    float r_ul = (i == 0) ? ((k == 0) ? 0.f : BIGF) : sb[c2][i - 1];
    float r_u  = (i == 0) ? BIGF : sb[c1][i - 1];
    float r_l  = sb[c1][i];
    float m = fminf(r_ul, fminf(r_u, r_l));
    float smin;
    if (m > 1e29f) smin = BIGF;
    else smin = m - 5.f * __logf(__expf((m - r_ul) * 0.2f) +
                                 __expf((m - r_u)  * 0.2f) +
                                 __expf((m - r_l)  * 0.2f));
    int j = k - i;
    float cur = BIGF;
    if (j >= 0 && j < LL) {
      size_t o = ((size_t)b * LL + i) * LL + j;
      cur = R[o] + smin;
      R[o] = cur;
      lmin = fminf(lmin, cur);
      lmax = fmaxf(lmax, cur);
    }
    sb[cc][i] = cur;
    __syncthreads();
    int t = c2; c2 = c1; c1 = cc; cc = t;
  }
  red[i] = lmin; __syncthreads();
  for (int s = 512; s > 0; s >>= 1) { if (i < s) red[i] = fminf(red[i], red[i + s]); __syncthreads(); }
  if (i == 0) rmin[b] = red[0];
  __syncthreads();
  red[i] = lmax; __syncthreads();
  for (int s = 512; s > 0; s >>= 1) { if (i < s) red[i] = fmaxf(red[i], red[i + s]); __syncthreads(); }
  if (i == 0) rmax[b] = red[0];
}

// ------- 4a) WMMA GEMM, 16x64 strip per wave: C = A*Y^T (+bias)(+relu) ------
__global__ void gemm4_f16_kernel(const _Float16* __restrict__ A, long long lda, long long strideA,
                                 const _Float16* __restrict__ Y, long long ldy, long long strideY,
                                 const float* __restrict__ bias,
                                 float* __restrict__ Cf, _Float16* __restrict__ Ch,
                                 long long ldc, long long strideC,
                                 int M, int N, int K, int relu) {
  int lane = threadIdx.x & 31, wave = threadIdx.x >> 5;
  int ng = N >> 6, mt = M >> 4;
  long long tile = (long long)blockIdx.x * 8 + wave;
  if (tile >= (long long)mt * ng) return;
  int i0 = (int)(tile / ng) << 4, j0 = (int)(tile % ng) << 6;
  int z = blockIdx.z;
  A += (size_t)z * strideA;
  Y += (size_t)z * strideY;
  v8f c[4];
#pragma unroll
  for (int g = 0; g < 4; g++)
    for (int i = 0; i < 8; i++) c[g][i] = 0.f;
  for (int k = 0; k < K; k += 32) {
    v16h a = load_frag_a(A, lda, i0, k, lane);
    if (k + 32 < K)   // -> global_prefetch_b8
      __builtin_prefetch(A + (size_t)(i0 + (lane & 15)) * lda + k + 32, 0, 1);
#pragma unroll
    for (int g = 0; g < 4; g++)
      c[g] = wmma32(a, load_frag_b(Y, ldy, j0 + 16 * g, k, lane), c[g]);
  }
  int nn = lane & 15, hi = lane >> 4;
#pragma unroll
  for (int g = 0; g < 4; g++) {
    int col = j0 + 16 * g + nn;
    float bc = bias ? bias[col] : 0.f;
#pragma unroll
    for (int r = 0; r < 8; r++) {
      int row = i0 + r + 8 * hi;
      float v = c[g][r] + bc;
      if (relu) v = fmaxf(v, 0.f);
      size_t o = (size_t)z * strideC + (size_t)row * ldc + col;
      if (Cf) Cf[o] = v;
      if (Ch) Ch[o] = (_Float16)v;
    }
  }
}

// ------- 4b) narrow WMMA GEMM (16x16/wave) with row scaling (final) ---------
__global__ void gemm_f16_kernel(const _Float16* __restrict__ A, long long lda,
                                const _Float16* __restrict__ Y, long long ldy,
                                const float* __restrict__ bias,
                                const float* __restrict__ rowScale,
                                float* __restrict__ Cf, long long ldc,
                                int M, int N, int K, int relu) {
  int lane = threadIdx.x & 31, wave = threadIdx.x >> 5;
  int nt = N >> 4, mt = M >> 4;
  long long tile = (long long)blockIdx.x * 8 + wave;
  if (tile >= (long long)mt * nt) return;
  int i0 = (int)(tile / nt) << 4, j0 = (int)(tile % nt) << 4;
  v8f c;
  for (int i = 0; i < 8; i++) c[i] = 0.f;
  for (int k = 0; k < K; k += 32)
    c = wmma32(load_frag_a(A, lda, i0, k, lane), load_frag_b(Y, ldy, j0, k, lane), c);
  int nn = lane & 15, hi = lane >> 4;
  float bc = bias ? bias[j0 + nn] : 0.f;
#pragma unroll
  for (int r = 0; r < 8; r++) {
    int row = i0 + r + 8 * hi;
    float v = c[r] + bc;
    if (relu) v = fmaxf(v, 0.f);
    if (rowScale) v *= rowScale[row];
    Cf[(size_t)row * ldc + j0 + nn] = v;
  }
}

// ------- 5) attention scores (WMMA 16x64/wave) + DTW bias, in place ---------
__global__ void scores_kernel(const _Float16* __restrict__ qkv16,
                              const float* __restrict__ mask,
                              const float* __restrict__ rmin,
                              const float* __restrict__ rmax,
                              float* __restrict__ R) {
  int lane = threadIdx.x & 31, wave = threadIdx.x >> 5;
  int b = blockIdx.z;
  int tile = blockIdx.x * 8 + wave;           // 64 i-tiles x 16 j-groups
  int i0 = (tile >> 4) << 4, j0 = (tile & 15) << 6;
  const _Float16* A = qkv16 + (size_t)b * LL * 384;        // q
  const _Float16* Y = A + 128;                             // k
  v8f c[4];
#pragma unroll
  for (int g = 0; g < 4; g++)
    for (int i = 0; i < 8; i++) c[g][i] = 0.f;
#pragma unroll
  for (int k = 0; k < DM; k += 32) {
    v16h a = load_frag_a(A, 384, i0, k, lane);
#pragma unroll
    for (int g = 0; g < 4; g++)
      c[g] = wmma32(a, load_frag_b(Y, 384, j0 + 16 * g, k, lane), c[g]);
  }
  int nn = lane & 15, hi = lane >> 4;
  float rmn = rmin[b];
  float sc  = 1.f / (rmax[b] - rmn);
#pragma unroll
  for (int g = 0; g < 4; g++) {
    int col = j0 + 16 * g + nn;
    float km = (mask[(size_t)b * LL + col] > 0.f) ? 0.f : -1e30f;
#pragma unroll
    for (int r = 0; r < 8; r++) {
      int row = i0 + r + 8 * hi;
      size_t o = ((size_t)b * LL + row) * LL + col;
      float biasv = (R[o] - rmn) * sc + 1.f;
      R[o] = c[g][r] * 0.08838834764831845f + biasv + km;   // 1/sqrt(128)
    }
  }
}

// ------- 6) row softmax over 1024 (async LDS row staging), emit f16 ---------
__global__ void softmax_kernel(const float* __restrict__ S,
                               _Float16* __restrict__ P) {
  __shared__ float red[256];
#if HAS_ASYNC_LDS
  __shared__ float xrow[LL];
#endif
  size_t row = blockIdx.x;
  const float* s = S + row * (size_t)LL;
  int t = threadIdx.x;
  float v[4];
#if HAS_ASYNC_LDS
  __builtin_amdgcn_global_load_async_to_lds_b128(
      (__attribute__((address_space(1))) v4i_t*)(s + t * 4),
      (__attribute__((address_space(3))) v4i_t*)(&xrow[t * 4]),
      0, 0);
#if __has_builtin(__builtin_amdgcn_s_wait_asynccnt)
  __builtin_amdgcn_s_wait_asynccnt(0);
#else
  asm volatile("s_wait_asynccnt 0" ::: "memory");
#endif
  __syncthreads();
#pragma unroll
  for (int i = 0; i < 4; i++) v[i] = xrow[t * 4 + i];
#else
#pragma unroll
  for (int i = 0; i < 4; i++) v[i] = s[t * 4 + i];
#endif
  float mx = -3e38f;
#pragma unroll
  for (int i = 0; i < 4; i++) mx = fmaxf(mx, v[i]);
  red[t] = mx; __syncthreads();
  for (int q = 128; q > 0; q >>= 1) { if (t < q) red[t] = fmaxf(red[t], red[t + q]); __syncthreads(); }
  mx = red[0]; __syncthreads();
  float e[4], sm = 0.f;
#pragma unroll
  for (int i = 0; i < 4; i++) { e[i] = __expf(v[i] - mx); sm += e[i]; }
  red[t] = sm; __syncthreads();
  for (int q = 128; q > 0; q >>= 1) { if (t < q) red[t] += red[t + q]; __syncthreads(); }
  float inv = 1.f / red[0];
#pragma unroll
  for (int i = 0; i < 4; i++) P[row * LL + t * 4 + i] = (_Float16)(e[i] * inv);
}

// ---------------- 7) small utility kernels ----------------------------------
__global__ void f2h_kernel(const float* __restrict__ s, _Float16* __restrict__ d, int n) {
  int i = blockIdx.x * 256 + threadIdx.x;
  if (i < n) d[i] = (_Float16)s[i];
}

__global__ void vt_kernel(const _Float16* __restrict__ qkv16, _Float16* __restrict__ vT) {
  int i = blockIdx.x * 256 + threadIdx.x;     // [b][d][l]
  if (i >= BB * DM * LL) return;
  int l = i & (LL - 1), dch = (i >> 10) & (DM - 1), b = i >> 17;
  vT[i] = qkv16[((size_t)b * LL + l) * 384 + 256 + dch];
}

__global__ void ln_kernel(const float* __restrict__ a, const float* __restrict__ b2,
                          const float* __restrict__ g, const float* __restrict__ bb,
                          float* __restrict__ outF, _Float16* __restrict__ outH) {
  __shared__ float rs[DM], rq[DM], st[2];
  size_t row = blockIdx.x;
  int d = threadIdx.x;
  float v = a[row * DM + d] + b2[row * DM + d];
  rs[d] = v; rq[d] = v * v; __syncthreads();
  for (int s = 64; s > 0; s >>= 1) {
    if (d < s) { rs[d] += rs[d + s]; rq[d] += rq[d + s]; }
    __syncthreads();
  }
  if (d == 0) {
    float m = rs[0] * (1.f / DM);
    st[0] = m;
    st[1] = rsqrtf(rq[0] * (1.f / DM) - m * m + 1e-5f);
  }
  __syncthreads();
  float o = (v - st[0]) * st[1] * g[d] + bb[d];
  if (outF) outF[row * DM + d] = o;
  outH[row * DM + d] = (_Float16)o;
}

__global__ void length_kernel(const float* __restrict__ mask, float* __restrict__ out) {
  __shared__ float red[256];
  int b = blockIdx.x, t = threadIdx.x;
  float s = 0.f;
  for (int i = t; i < LL; i += 256) s += mask[(size_t)b * LL + i];
  red[t] = s; __syncthreads();
  for (int q = 128; q > 0; q >>= 1) { if (t < q) red[t] += red[t + q]; __syncthreads(); }
  if (t == 0) out[b] = red[0];
}

// ---------------------------------------------------------------------------
extern "C" void kernel_launch(void* const* d_in, const int* in_sizes, int n_in,
                              void* d_out, int out_size, void* d_ws, size_t ws_size,
                              hipStream_t stream) {
  (void)in_sizes; (void)n_in; (void)out_size; (void)ws_size;
  const float* x      = (const float*)d_in[0];
  const float* mask   = (const float*)d_in[1];
  const float* conv_w = (const float*)d_in[2];
  const float* conv_b = (const float*)d_in[3];
  const float* w_in   = (const float*)d_in[4];
  const float* b_in   = (const float*)d_in[5];
  const float* w_out  = (const float*)d_in[6];
  const float* b_out  = (const float*)d_in[7];
  const float* w_ff1  = (const float*)d_in[8];
  const float* b_ff1  = (const float*)d_in[9];
  const float* w_ff2  = (const float*)d_in[10];
  const float* b_ff2  = (const float*)d_in[11];
  const float* ln1_g  = (const float*)d_in[12];
  const float* ln1_b  = (const float*)d_in[13];
  const float* ln2_g  = (const float*)d_in[14];
  const float* ln2_b  = (const float*)d_in[15];
  const float* w_lin  = (const float*)d_in[16];
  float* out = (float*)d_out;

  // workspace carve-up (256B aligned)
  char* w = (char*)d_ws;
  size_t off = 0;
  auto alloc = [&](size_t bytes) -> void* {
    void* p = w + off;
    off = (off + bytes + 255) & ~(size_t)255;
    return p;
  };
  const size_t ML = (size_t)BB * LL;            // 32768 rows
  float*    Rbuf  = (float*)   alloc((size_t)BB * LL * LL * 4);  // D2 -> R -> scores
  _Float16* P16   = (_Float16*)alloc((size_t)BB * LL * LL * 2);  // attn probs
  float*    hbuf  = (float*)   alloc(ML * DM * 4);
  _Float16* h16   = (_Float16*)alloc(ML * DM * 2);
  float*    norms = (float*)   alloc(ML * 4);
  float*    rmn   = (float*)   alloc(BB * 4);
  float*    rmx   = (float*)   alloc(BB * 4);
  _Float16* qkv16 = (_Float16*)alloc(ML * 384 * 2);
  _Float16* vT16  = (_Float16*)alloc((size_t)BB * DM * LL * 2);
  _Float16* sa16  = (_Float16*)alloc(ML * DM * 2);
  float*    proj  = (float*)   alloc(ML * DM * 4);
  float*    h1f   = (float*)   alloc(ML * DM * 4);
  _Float16* h1h   = (_Float16*)alloc(ML * DM * 2);
  _Float16* ff16  = (_Float16*)alloc(ML * DM * 2);
  float*    ff2f  = (float*)   alloc(ML * DM * 4);
  _Float16* h2h   = (_Float16*)alloc(ML * DM * 2);
  _Float16* win16 = (_Float16*)alloc(384 * DM * 2);
  _Float16* wout16= (_Float16*)alloc(DM * DM * 2);
  _Float16* wff116= (_Float16*)alloc(DM * DM * 2);
  _Float16* wff216= (_Float16*)alloc(DM * DM * 2);
  _Float16* wlin16= (_Float16*)alloc(16 * DM * 2);

  // weights -> f16
  f2h_kernel<<<(384 * DM + 255) / 256, 256, 0, stream>>>(w_in,  win16,  384 * DM);
  f2h_kernel<<<(DM * DM + 255) / 256, 256, 0, stream>>>(w_out,  wout16, DM * DM);
  f2h_kernel<<<(DM * DM + 255) / 256, 256, 0, stream>>>(w_ff1,  wff116, DM * DM);
  f2h_kernel<<<(DM * DM + 255) / 256, 256, 0, stream>>>(w_ff2,  wff216, DM * DM);
  f2h_kernel<<<(16 * DM + 255) / 256, 256, 0, stream>>>(w_lin,  wlin16, 16 * DM);

  // 1) conv + pool + relu (+ mask, + row norms)
  conv_kernel<<<BB * LL, DM, 0, stream>>>(x, conv_w, conv_b, mask, hbuf, h16, norms);

  // 2) pairwise squared distances (WMMA, 16x64 per wave)
  dist_kernel<<<dim3(128, 1, BB), 256, 0, stream>>>(h16, norms, Rbuf);

  // 3) soft-DTW wavefront (in-place R, per-batch min/max)
  softdtw_kernel<<<BB, LL, 0, stream>>>(Rbuf, rmn, rmx);

  // 4) qkv = h @ w_in^T + b_in  (M=32768, N=384, K=128) -> f16
  gemm4_f16_kernel<<<(2048 * 6) / 8, 256, 0, stream>>>(
      h16, DM, 0, win16, DM, 0, b_in, nullptr, qkv16, 384, 0,
      (int)ML, 384, DM, 0);

  // 5) v transposed per batch: vT[b][d][l]
  vt_kernel<<<(BB * DM * LL) / 256, 256, 0, stream>>>(qkv16, vT16);

  // 6) scores = q k^T / sqrt(d) + dtw_bias (+key mask), in place into Rbuf
  scores_kernel<<<dim3(128, 1, BB), 256, 0, stream>>>(qkv16, mask, rmn, rmx, Rbuf);

  // 7) softmax rows -> P16
  softmax_kernel<<<(int)ML, 256, 0, stream>>>(Rbuf, P16);

  // 8) sa = P @ V  (batched: M=1024, N=128, K=1024) -> f16
  gemm4_f16_kernel<<<dim3((64 * 2) / 8, 1, BB), 256, 0, stream>>>(
      P16, LL, (long long)LL * LL, vT16, LL, (long long)DM * LL,
      nullptr, nullptr, sa16, DM, (long long)LL * DM,
      LL, DM, LL, 0);

  // 9) out-projection: proj = sa @ w_out^T + b_out (f32)
  gemm4_f16_kernel<<<(2048 * 2) / 8, 256, 0, stream>>>(
      sa16, DM, 0, wout16, DM, 0, b_out, proj, nullptr, DM, 0,
      (int)ML, DM, DM, 0);

  // 10) LN1(h + proj)
  ln_kernel<<<(int)ML, DM, 0, stream>>>(hbuf, proj, ln1_g, ln1_b, h1f, h1h);

  // 11) ff = relu(h1 @ w_ff1^T + b_ff1) -> f16
  gemm4_f16_kernel<<<(2048 * 2) / 8, 256, 0, stream>>>(
      h1h, DM, 0, wff116, DM, 0, b_ff1, nullptr, ff16, DM, 0,
      (int)ML, DM, DM, 1);

  // 12) ff2 = ff @ w_ff2^T + b_ff2 (f32)
  gemm4_f16_kernel<<<(2048 * 2) / 8, 256, 0, stream>>>(
      ff16, DM, 0, wff216, DM, 0, b_ff2, ff2f, nullptr, DM, 0,
      (int)ML, DM, DM, 0);

  // 13) LN2(h1 + ff2) -> h2 (f16 only)
  ln_kernel<<<(int)ML, DM, 0, stream>>>(h1f, ff2f, ln2_g, ln2_b, nullptr, h2h);

  // 14) out = (h2 @ w_lin^T) * mask   (M=32768, N=16, K=128) -> d_out
  gemm_f16_kernel<<<2048 / 8, 256, 0, stream>>>(
      h2h, DM, wlin16, DM, nullptr, mask, out, 16, (int)ML, 16, DM, 0);

  // 15) length = sum(mask, axis=1) -> tail of d_out
  length_kernel<<<BB, 256, 0, stream>>>(mask, out + (size_t)ML * 16);
}
